// DEBertAAttentionHead_81080392614801
// MI455X (gfx1250) — matrix operand order repaired and checked by reading, not compiled, not run on hardware
//
#include <hip/hip_runtime.h>

// ---------------------------------------------------------------------------
// DEBertAAttentionHead on MI455X (gfx1250, wave32, WMMA 16x16x32 bf16)
//   B=8, S=2048, E=768, H=64, fp32 in/out; bf16 WMMA operands, f32 accum.
//   Kernel 1: fused q/k/v projections -> bf16 q[S,64], k[S,64], vT[64,S]
//   Kernel 2: split-K flash-attention: 4 waves per 16-query tile, each wave
//             owns a 512-key quarter with its own online softmax; partials
//             merged through LDS (log-sum-exp combine) after the barrier.
// ---------------------------------------------------------------------------

typedef __attribute__((ext_vector_type(16))) __bf16 v16bf;
typedef __attribute__((ext_vector_type(8)))  float  v8f;

#define S_LEN 2048
#define BATCH 8
#define NEMB  768
#define HEAD  64
#define ROWS  (BATCH * S_LEN)   // 16384 flattened rows
#define KSPLIT 4                // key-dimension split (waves per query tile)

__device__ __forceinline__ __bf16 us2bf(unsigned short u) {
  union { unsigned short s; __bf16 b; } c; c.s = u; return c.b;
}

__device__ __forceinline__ unsigned short f2bf_bits(float f) {
  unsigned int x = __float_as_uint(f);
  unsigned int r = x + 0x7FFFu + ((x >> 16) & 1u);   // round-to-nearest-even
  return (unsigned short)(r >> 16);
}

// 16 contiguous bf16 (two uint4 loads) -> v16bf fragment
__device__ __forceinline__ v16bf unpack16(uint4 u0, uint4 u1) {
  v16bf r;
  unsigned int w0 = u0.x, w1 = u0.y, w2 = u0.z, w3 = u0.w;
  unsigned int w4 = u1.x, w5 = u1.y, w6 = u1.z, w7 = u1.w;
  r[0]  = us2bf((unsigned short)(w0 & 0xFFFF)); r[1]  = us2bf((unsigned short)(w0 >> 16));
  r[2]  = us2bf((unsigned short)(w1 & 0xFFFF)); r[3]  = us2bf((unsigned short)(w1 >> 16));
  r[4]  = us2bf((unsigned short)(w2 & 0xFFFF)); r[5]  = us2bf((unsigned short)(w2 >> 16));
  r[6]  = us2bf((unsigned short)(w3 & 0xFFFF)); r[7]  = us2bf((unsigned short)(w3 >> 16));
  r[8]  = us2bf((unsigned short)(w4 & 0xFFFF)); r[9]  = us2bf((unsigned short)(w4 >> 16));
  r[10] = us2bf((unsigned short)(w5 & 0xFFFF)); r[11] = us2bf((unsigned short)(w5 >> 16));
  r[12] = us2bf((unsigned short)(w6 & 0xFFFF)); r[13] = us2bf((unsigned short)(w6 >> 16));
  r[14] = us2bf((unsigned short)(w7 & 0xFFFF)); r[15] = us2bf((unsigned short)(w7 >> 16));
  return r;
}

// ---------------------------------------------------------------------------
// Kernel 1: q = I@Wq+bq ; k = x@Wk+bk ; v = x@Wv+bv   (bf16 outputs)
// 12 waves / block; wave w: mat = w/4 (0=q,1=k,2=v), col-tile = w%4.
// Each wave: one 16x16 output tile, K=768 as 24 chained bf16 WMMAs.
// A frag (16x32 MxK): lane L: M = L%16; elem j -> K = (j<8? j : 16+j-8) + 8*(L/16)
// B frag (32x16 KxN): lane L: N = L%16; elem j -> K = (j<8? j : 16+j-8) + 8*(L/16)
// D frag (16x16):     lane L: N = L%16; vgpr r -> M = r + 8*(L/16)
// ---------------------------------------------------------------------------
__global__ __launch_bounds__(384) void qkv_proj_kernel(
    const float* __restrict__ I, const float* __restrict__ X,
    const float* __restrict__ Wq, const float* __restrict__ bq,
    const float* __restrict__ Wk, const float* __restrict__ bk,
    const float* __restrict__ Wv, const float* __restrict__ bv,
    unsigned short* __restrict__ qbf, unsigned short* __restrict__ kbf,
    unsigned short* __restrict__ vT)
{
  const int wave = threadIdx.x >> 5;
  const int lane = threadIdx.x & 31;
  const int hl   = lane >> 4;        // lane half
  const int ln   = lane & 15;
  const int mat  = wave >> 2;        // 0=q, 1=k, 2=v
  const int nt   = wave & 3;         // 16-wide head-column tile
  const int rowbase = blockIdx.x * 16;

  const float* src  = (mat == 0) ? I  : X;
  const float* W    = (mat == 0) ? Wq : (mat == 1) ? Wk : Wv;
  const float* bias = (mat == 0) ? bq : (mat == 1) ? bk : bv;

  v8f acc = {};
  const float* arow = src + (size_t)(rowbase + ln) * NEMB + hl * 8;
  const float* wcol = W + (size_t)(hl * 8) * HEAD + nt * 16 + ln;

  for (int kb = 0; kb < NEMB; kb += 32) {
    float4 a0 = *(const float4*)(arow + kb + 0);
    float4 a1 = *(const float4*)(arow + kb + 4);
    float4 a2 = *(const float4*)(arow + kb + 16);
    float4 a3 = *(const float4*)(arow + kb + 20);
    v16bf a, bfr;
    a[0]  = (__bf16)a0.x; a[1]  = (__bf16)a0.y; a[2]  = (__bf16)a0.z; a[3]  = (__bf16)a0.w;
    a[4]  = (__bf16)a1.x; a[5]  = (__bf16)a1.y; a[6]  = (__bf16)a1.z; a[7]  = (__bf16)a1.w;
    a[8]  = (__bf16)a2.x; a[9]  = (__bf16)a2.y; a[10] = (__bf16)a2.z; a[11] = (__bf16)a2.w;
    a[12] = (__bf16)a3.x; a[13] = (__bf16)a3.y; a[14] = (__bf16)a3.z; a[15] = (__bf16)a3.w;
    const float* wp = wcol + (size_t)kb * HEAD;
#pragma unroll
    for (int j = 0; j < 16; ++j) bfr[j] = (__bf16)wp[(size_t)(j < 8 ? j : j + 8) * HEAD];
    acc = __builtin_amdgcn_wmma_f32_16x16x32_bf16(false, a, false, bfr,
                                                  (short)0, acc, false, false);
  }

  float bn = bias[nt * 16 + ln];
#pragma unroll
  for (int r = 0; r < 8; ++r) acc[r] += bn;

  if (mat < 2) {
    // q/k row-major [ROWS, 64] bf16 (matches A-operand / B-operand loads later)
    unsigned short* dst = (mat == 0) ? qbf : kbf;
#pragma unroll
    for (int r = 0; r < 8; ++r) {
      int row = rowbase + r + 8 * hl;
      dst[(size_t)row * HEAD + nt * 16 + ln] = f2bf_bits(acc[r]);
    }
  } else {
    // v transposed [64, ROWS] bf16: 8 contiguous s-values -> one 16B store
    int h = nt * 16 + ln;
    uint4 pk;
    pk.x = (unsigned int)f2bf_bits(acc[0]) | ((unsigned int)f2bf_bits(acc[1]) << 16);
    pk.y = (unsigned int)f2bf_bits(acc[2]) | ((unsigned int)f2bf_bits(acc[3]) << 16);
    pk.z = (unsigned int)f2bf_bits(acc[4]) | ((unsigned int)f2bf_bits(acc[5]) << 16);
    pk.w = (unsigned int)f2bf_bits(acc[6]) | ((unsigned int)f2bf_bits(acc[7]) << 16);
    *(uint4*)(vT + (size_t)h * ROWS + rowbase + 8 * hl) = pk;
  }
}

// ---------------------------------------------------------------------------
// Kernel 2: split-K flash attention.
// Block = 128 threads = 4 waves, one 16-query tile per block.
// Wave w streams keys [w*512, (w+1)*512) with private online-softmax state.
// S^T tile = K_tile(A, 16key x 64emb as 2 chained K=32) * Q^T(B): each
// softmax row (one query) lives in lanes {n, n+16} -> cheap reductions.
// Context^T accumulated as V^T(A) * P^T(B), 4 head tiles of 16.
// Partials merged via LDS with log-sum-exp combine after __syncthreads().
// ---------------------------------------------------------------------------
__global__ __launch_bounds__(128) void attn_kernel(
    const int* __restrict__ mask,
    const unsigned short* __restrict__ qbf,
    const unsigned short* __restrict__ kbf,
    const unsigned short* __restrict__ vT,
    float* __restrict__ out)
{
  __shared__ float lds_acc[(KSPLIT - 1) * 32 * 32];  // waves 1..3: 32 f32/lane
  __shared__ float lds_ml[(KSPLIT - 1) * 32 * 2];    // waves 1..3: m,l per lane

  const int wave = threadIdx.x >> 5;   // key-split index 0..3
  const int lane = threadIdx.x & 31;
  const int hl   = lane >> 4;
  const int ln   = lane & 15;
  const int b    = blockIdx.y;
  const int qbase = blockIdx.x * 16;
  const float rscale = 0.036084391824351615f;   // 1/sqrt(768)

  // Q as B-operand, two K=32 chains: elem j -> h = (j<8? j : 16+j-8) + 8*hl + 32*c
  const unsigned short* qrow =
      qbf + (size_t)(b * S_LEN + qbase + ln) * HEAD + 8 * hl;
  v16bf qf[2];
#pragma unroll
  for (int c = 0; c < 2; ++c) {
    uint4 u0 = *(const uint4*)(qrow + 32 * c);
    uint4 u1 = *(const uint4*)(qrow + 32 * c + 16);
    qf[c] = unpack16(u0, u1);
  }

  float m_run = -3.0e38f, l_run = 0.0f;
  v8f acc[4] = {v8f{}, v8f{}, v8f{}, v8f{}};

  const unsigned short* kmat = kbf + (size_t)b * S_LEN * HEAD;
  const int* mrow = mask + ((size_t)b * S_LEN + qbase + ln) * S_LEN + 8 * hl;

  const int kbeg = wave * (S_LEN / KSPLIT);
  const int kend = kbeg + (S_LEN / KSPLIT);

  for (int kb = kbeg; kb < kend; kb += 32) {
    float p[2][8];
    float cmax = -3.0e38f;

#pragma unroll
    for (int t = 0; t < 2; ++t) {
      v8f sD = {};
#pragma unroll
      for (int c = 0; c < 2; ++c) {
        // K rows as A-operand: M = key = kb+16t+ln;
        // elem j -> h = (j<8? j : 16+j-8) + 8*hl + 32*c
        const unsigned short* kr =
            kmat + (size_t)(kb + 16 * t + ln) * HEAD + 8 * hl + 32 * c;
        uint4 u0 = *(const uint4*)(kr);
        uint4 u1 = *(const uint4*)(kr + 16);
        v16bf a = unpack16(u0, u1);
        sD = __builtin_amdgcn_wmma_f32_16x16x32_bf16(false, a, false, qf[c],
                                                     (short)0, sD, false, false);
      }
      // lane holds query q=ln, keys kb+16t + r + 8*hl (r = vgpr index)
      const int* mp = mrow + kb + 16 * t;
      int4 mm0 = *(const int4*)(mp);
      int4 mm1 = *(const int4*)(mp + 4);
      float s0 = (mm0.x == 0) ? -1.0e9f : sD[0] * rscale;
      float s1 = (mm0.y == 0) ? -1.0e9f : sD[1] * rscale;
      float s2 = (mm0.z == 0) ? -1.0e9f : sD[2] * rscale;
      float s3 = (mm0.w == 0) ? -1.0e9f : sD[3] * rscale;
      float s4 = (mm1.x == 0) ? -1.0e9f : sD[4] * rscale;
      float s5 = (mm1.y == 0) ? -1.0e9f : sD[5] * rscale;
      float s6 = (mm1.z == 0) ? -1.0e9f : sD[6] * rscale;
      float s7 = (mm1.w == 0) ? -1.0e9f : sD[7] * rscale;
      p[t][0] = s0; p[t][1] = s1; p[t][2] = s2; p[t][3] = s3;
      p[t][4] = s4; p[t][5] = s5; p[t][6] = s6; p[t][7] = s7;
      cmax = fmaxf(cmax, fmaxf(fmaxf(fmaxf(s0, s1), fmaxf(s2, s3)),
                               fmaxf(fmaxf(s4, s5), fmaxf(s6, s7))));
    }
    // one lane-pair exchange completes the 32-key row max
    cmax = fmaxf(cmax, __shfl_xor(cmax, 16, 32));

    float mnew = fmaxf(m_run, cmax);
    float scale_old = __expf(m_run - mnew);
    float psum = 0.0f;
#pragma unroll
    for (int t = 0; t < 2; ++t)
#pragma unroll
      for (int r = 0; r < 8; ++r) {
        p[t][r] = __expf(p[t][r] - mnew);
        psum += p[t][r];
      }
    psum += __shfl_xor(psum, 16, 32);
    l_run = l_run * scale_old + psum;
    m_run = mnew;

#pragma unroll
    for (int ht = 0; ht < 4; ++ht)
#pragma unroll
      for (int r = 0; r < 8; ++r) acc[ht][r] *= scale_old;

    // Build P^T as B-operand (32key x 16query): lane-pair swap per row.
    // half0 lane needs keys kb+0..15, half1 lane keys kb+16..31.
    v16bf pB;
#pragma unroll
    for (int r = 0; r < 8; ++r) {
      float o0 = __shfl_xor(p[0][r], 16, 32);
      float o1 = __shfl_xor(p[1][r], 16, 32);
      float e_lo = hl ? o1 : p[0][r];   // elem j = r
      float e_hi = hl ? p[1][r] : o0;   // elem j = 8 + r
      pB[r]     = (__bf16)e_lo;
      pB[8 + r] = (__bf16)e_hi;
    }

    // C^T += V^T(A: 16h x 32key) * P^T(B)
#pragma unroll
    for (int ht = 0; ht < 4; ++ht) {
      const unsigned short* vr =
          vT + (size_t)(ht * 16 + ln) * ROWS + b * S_LEN + kb + 16 * hl;
      uint4 u0 = *(const uint4*)(vr);
      uint4 u1 = *(const uint4*)(vr + 8);
      v16bf a = unpack16(u0, u1);
      acc[ht] = __builtin_amdgcn_wmma_f32_16x16x32_bf16(false, a, false, pB,
                                                        (short)0, acc[ht], false, false);
    }
  }

  // ---- split-K merge through LDS (all WMMA issued before any divergence) ---
  if (wave != 0) {
    int base = ((wave - 1) * 32 + lane) * 32;
#pragma unroll
    for (int ht = 0; ht < 4; ++ht)
#pragma unroll
      for (int r = 0; r < 8; ++r) lds_acc[base + ht * 8 + r] = acc[ht][r];
    lds_ml[((wave - 1) * 32 + lane) * 2 + 0] = m_run;
    lds_ml[((wave - 1) * 32 + lane) * 2 + 1] = l_run;
  }
  __syncthreads();
  if (wave != 0) return;

  float mw0 = lds_ml[(0 * 32 + lane) * 2 + 0], lw0 = lds_ml[(0 * 32 + lane) * 2 + 1];
  float mw1 = lds_ml[(1 * 32 + lane) * 2 + 0], lw1 = lds_ml[(1 * 32 + lane) * 2 + 1];
  float mw2 = lds_ml[(2 * 32 + lane) * 2 + 0], lw2 = lds_ml[(2 * 32 + lane) * 2 + 1];
  float m_tot = fmaxf(fmaxf(m_run, mw0), fmaxf(mw1, mw2));

  float sc  = __expf(m_run - m_tot);
  float sc0 = __expf(mw0 - m_tot);
  float sc1 = __expf(mw1 - m_tot);
  float sc2 = __expf(mw2 - m_tot);
  float l_tot = l_run * sc + lw0 * sc0 + lw1 * sc1 + lw2 * sc2;

  int b0 = (0 * 32 + lane) * 32;
  int b1 = (1 * 32 + lane) * 32;
  int b2 = (2 * 32 + lane) * 32;
#pragma unroll
  for (int ht = 0; ht < 4; ++ht)
#pragma unroll
    for (int r = 0; r < 8; ++r) {
      float v = acc[ht][r] * sc;
      v += lds_acc[b0 + ht * 8 + r] * sc0;
      v += lds_acc[b1 + ht * 8 + r] * sc1;
      v += lds_acc[b2 + ht * 8 + r] * sc2;
      acc[ht][r] = v;
    }

  // epilogue: divide by softmax denom; C^T D-layout -> contiguous f32 stores
  float inv = 1.0f / l_tot;
  float* orow = out + (size_t)(b * S_LEN + qbase + ln) * HEAD + 8 * hl;
#pragma unroll
  for (int ht = 0; ht < 4; ++ht) {
    float4 o0 = make_float4(acc[ht][0] * inv, acc[ht][1] * inv,
                            acc[ht][2] * inv, acc[ht][3] * inv);
    float4 o1 = make_float4(acc[ht][4] * inv, acc[ht][5] * inv,
                            acc[ht][6] * inv, acc[ht][7] * inv);
    *(float4*)(orow + 16 * ht)     = o0;
    *(float4*)(orow + 16 * ht + 4) = o1;
  }
}

// ---------------------------------------------------------------------------
extern "C" void kernel_launch(void* const* d_in, const int* in_sizes, int n_in,
                              void* d_out, int out_size, void* d_ws, size_t ws_size,
                              hipStream_t stream) {
  const float* I  = (const float*)d_in[0];
  const float* X  = (const float*)d_in[1];
  const int* mask = (const int*)d_in[2];
  const float* Wq = (const float*)d_in[3];
  const float* bq = (const float*)d_in[4];
  const float* Wk = (const float*)d_in[5];
  const float* bk = (const float*)d_in[6];
  const float* Wv = (const float*)d_in[7];
  const float* bv = (const float*)d_in[8];
  float* out = (float*)d_out;

  // workspace: q[16384,64] + k[16384,64] + vT[64,16384], bf16 -> 6 MB
  unsigned short* qbf = (unsigned short*)d_ws;
  unsigned short* kbf = qbf + (size_t)ROWS * HEAD;
  unsigned short* vTb = kbf + (size_t)ROWS * HEAD;

  qkv_proj_kernel<<<ROWS / 16, 384, 0, stream>>>(I, X, Wq, bq, Wk, bk, Wv, bv,
                                                 qbf, kbf, vTb);
  attn_kernel<<<dim3(S_LEN / 16, BATCH), 128, 0, stream>>>(mask, qbf, kbf,
                                                           vTb, out);
}
